// ReadSetClassifier_4337916969088
// MI455X (gfx1250) — matrix operand order repaired and checked by reading, not compiled, not run on hardware
//
#include <hip/hip_runtime.h>
#include <hip/hip_bf16.h>
#include <math.h>

// ---------------------------------------------------------------------------
// Problem constants (match reference)
#define V_SEG   8192
#define NREF    262144
#define NALT    131072
#define NTOT    (NREF + NALT)
#define F_READ  11
#define F_INFO  9
#define H_DIM   256
#define MAX_ALT 10

// ---------------------------------------------------------------------------
// WMMA types (gfx1250, wave32)
typedef __attribute__((ext_vector_type(16))) __bf16 v16bf;
typedef __attribute__((ext_vector_type(8)))  float  v8f;

union AFrag { v16bf v; uint4 q[2]; };

__device__ __forceinline__ unsigned short f2bf(float f) {
    unsigned int u = __float_as_uint(f);
    unsigned int r = u + 0x7FFFu + ((u >> 16) & 1u);   // round-to-nearest-even
    return (unsigned short)(r >> 16);
}
__device__ __forceinline__ float bf2f(unsigned short h) {
    return __uint_as_float(((unsigned int)h) << 16);
}
__device__ __forceinline__ float lrelu(float x) { return x > 0.f ? x : 0.01f * x; }
__device__ __forceinline__ float sigmoidf(float x) { return 1.f / (1.f + expf(-x)); }

#define WMMA_BF16(A, B, C) \
    __builtin_amdgcn_wmma_f32_16x16x32_bf16(false, (A), false, (B), (short)0, (C), false, false)

// ---------------------------------------------------------------------------
// Zero a float region
__global__ void zero_kernel(float* p, int n) {
    for (int i = blockIdx.x * blockDim.x + threadIdx.x; i < n; i += gridDim.x * blockDim.x)
        p[i] = 0.f;
}

// Transpose f32 [K][N] -> bf16 [N][K]
__global__ void tw_kernel(const float* __restrict__ src, unsigned short* __restrict__ dst,
                          int K, int N) {
    int idx = blockIdx.x * blockDim.x + threadIdx.x;
    if (idx >= K * N) return;
    int k = idx / N, n = idx % N;
    dst[(size_t)n * K + k] = f2bf(src[idx]);
}

// Per-segment read counts (as float, matching reference's float counts)
__global__ void count_kernel(const int* __restrict__ seg, float* __restrict__ cnt, int n) {
    int i = blockIdx.x * blockDim.x + threadIdx.x;
    if (i < n) atomicAdd(&cnt[seg[i]], 1.0f);
}

// ---------------------------------------------------------------------------
// Fused phi MLP (11 -> 256 lrelu -> 256 sigmoid) + segmented sum accumulation.
// 128 rows per block, 128 threads (4 waves); each wave owns TWO 16-row tiles
// (32 rows) so every B fragment of phi_W1 is reused by two WMMAs -> halves
// L2 weight traffic on the dominant 51.5 GFLOP GEMM.
#define PHI_ROWS 128
__global__ __launch_bounds__(128)
void phi_kernel(const float* __restrict__ reads,
                const int*   __restrict__ ref_seg,
                const int*   __restrict__ alt_seg,
                const float* __restrict__ W0,   // 11 x 256
                const float* __restrict__ b0,   // 256
                const unsigned short* __restrict__ W1t, // 256 x 256 (N-major bf16)
                const float* __restrict__ b1,   // 256
                float* __restrict__ ref_sums,   // V x 256
                float* __restrict__ alt_sums)   // V x 256
{
    extern __shared__ char smem[];
    unsigned short* hbuf = (unsigned short*)smem;                       // 128x256 bf16 = 64 KB
    float* rbuf  = (float*)(smem + PHI_ROWS * H_DIM * 2);               // 128x11 f32
    int*   segbuf = (int*)(smem + PHI_ROWS * H_DIM * 2 + PHI_ROWS * F_READ * 4);
    int*   refbuf = segbuf + PHI_ROWS;

    const int tid = threadIdx.x;
    const int rowbase = blockIdx.x * PHI_ROWS;

    // stage reads + segment metadata
    for (int i = tid; i < PHI_ROWS * F_READ; i += 128)
        rbuf[i] = reads[(size_t)rowbase * F_READ + i];
    if (tid < PHI_ROWS) {
        int g = rowbase + tid;
        if (g < NREF) { segbuf[tid] = ref_seg[g];        refbuf[tid] = 1; }
        else          { segbuf[tid] = alt_seg[g - NREF]; refbuf[tid] = 0; }
    }
    __syncthreads();

    // layer 0: K=11, scalar FMA; each thread handles 2 columns for all rows
    #pragma unroll
    for (int cc = 0; cc < 2; ++cc) {
        int c = tid + cc * 128;
        float w[F_READ];
        #pragma unroll
        for (int k = 0; k < F_READ; ++k) w[k] = W0[k * H_DIM + c];
        float bb = b0[c];
        for (int r = 0; r < PHI_ROWS; ++r) {
            float s = bb;
            #pragma unroll
            for (int k = 0; k < F_READ; ++k) s += rbuf[r * F_READ + k] * w[k];
            hbuf[r * H_DIM + c] = f2bf(lrelu(s));
        }
    }
    __syncthreads();

    // layer 1: 256x256 bf16 WMMA, two M-tiles per wave, sigmoid, atomic segsum
    const int wave  = tid >> 5;
    const int lane  = tid & 31;
    const int wrow  = wave * 32;                  // 32 rows per wave
    const int arow0 = wrow + (lane & 15);
    const int arow1 = arow0 + 16;
    const int koffA = (lane < 16) ? 0 : 16;
    const int koffB = (lane < 16) ? 0 : 16;
    const int ncol  = lane & 15;
    const int mbase = (lane < 16) ? 0 : 8;

    // hoist all A fragments for both tiles into registers (16 frags = 128 VGPRs)
    AFrag a0[8], a1[8];
    #pragma unroll
    for (int kf = 0; kf < 8; ++kf) {
        int kk = kf * 32;
        a0[kf].q[0] = *(const uint4*)(&hbuf[arow0 * H_DIM + kk + koffA]);
        a0[kf].q[1] = *(const uint4*)(&hbuf[arow0 * H_DIM + kk + 8 + koffA]);
        a1[kf].q[0] = *(const uint4*)(&hbuf[arow1 * H_DIM + kk + koffA]);
        a1[kf].q[1] = *(const uint4*)(&hbuf[arow1 * H_DIM + kk + 8 + koffA]);
    }

    for (int nt = 0; nt < 16; ++nt) {
        int n = nt * 16 + ncol;
        const unsigned short* Bn = W1t + (size_t)n * H_DIM;
        if (nt < 15)                                   // prime next weight row
            __builtin_prefetch(Bn + H_DIM * 16, 0, 1); // -> global_prefetch_b8
        v8f acc0 = {}, acc1 = {};
        #pragma unroll
        for (int kf = 0; kf < 8; ++kf) {
            v16bf b = *(const v16bf*)(Bn + kf * 32 + koffB);
            acc0 = WMMA_BF16(a0[kf].v, b, acc0);
            acc1 = WMMA_BF16(a1[kf].v, b, acc1);
        }
        float bv = b1[n];
        #pragma unroll
        for (int r = 0; r < 8; ++r) {
            {
                float v = sigmoidf(acc0[r] + bv);
                int lr = wrow + mbase + r;
                float* dst = refbuf[lr] ? ref_sums : alt_sums;
                atomicAdd(dst + (size_t)segbuf[lr] * H_DIM + n, v);
            }
            {
                float v = sigmoidf(acc1[r] + bv);
                int lr = wrow + 16 + mbase + r;
                float* dst = refbuf[lr] ? ref_sums : alt_sums;
                atomicAdd(dst + (size_t)segbuf[lr] * H_DIM + n, v);
            }
        }
    }
}

// ---------------------------------------------------------------------------
// Build concat input X = [ref_mean | alt_mean | sigmoid(info@om_W0+om_b0)] (bf16)
__global__ void buildx_kernel(const float* __restrict__ ref_sums,
                              const float* __restrict__ alt_sums,
                              const float* __restrict__ ref_cnt,
                              const float* __restrict__ alt_cnt,
                              const float* __restrict__ info,
                              const float* __restrict__ omW0,
                              const float* __restrict__ omb0,
                              unsigned short* __restrict__ X)
{
    int idx = blockIdx.x * blockDim.x + threadIdx.x;   // V*256 threads
    if (idx >= V_SEG * H_DIM) return;
    int v = idx >> 8, h = idx & 255;
    float rm = ref_sums[idx] / fmaxf(ref_cnt[v], 1.f);
    float am = alt_sums[idx] / fmaxf(alt_cnt[v], 1.f);
    float s = omb0[h];
    #pragma unroll
    for (int k = 0; k < F_INFO; ++k) s += info[v * F_INFO + k] * omW0[k * H_DIM + h];
    s = sigmoidf(s);
    size_t base = (size_t)v * (3 * H_DIM);
    X[base + h]             = f2bf(rm);
    X[base + H_DIM + h]     = f2bf(am);
    X[base + 2 * H_DIM + h] = f2bf(s);
}

// ---------------------------------------------------------------------------
// Generic bf16 WMMA GEMM: C[M x 256] = act(A[M x K] @ B[K x 256] + bias)
// Bt is N-major (256 x K). ACT: 0 = none, 1 = leaky relu.
// 256 threads = 8 waves; each wave owns two 16-row tiles -> 256 rows/block.
template <int ACT>
__global__ __launch_bounds__(256)
void gemm_kernel(const unsigned short* __restrict__ A,
                 const unsigned short* __restrict__ Bt,
                 const float* __restrict__ bias,
                 unsigned short* __restrict__ C,
                 int K)
{
    const int wave  = threadIdx.x >> 5;
    const int lane  = threadIdx.x & 31;
    const int rowb  = blockIdx.x * 256 + wave * 32;
    const int koffA = (lane < 16) ? 0 : 16;
    const int koffB = (lane < 16) ? 0 : 16;
    const int ncol  = lane & 15;
    const int mbase = (lane < 16) ? 0 : 8;
    const unsigned short* Arow0 = A + (size_t)(rowb + (lane & 15)) * K;
    const unsigned short* Arow1 = Arow0 + (size_t)16 * K;

    for (int nt = 0; nt < 16; ++nt) {
        int n = nt * 16 + ncol;
        const unsigned short* Bn = Bt + (size_t)n * K;
        if (nt < 15)
            __builtin_prefetch(Bn + (size_t)16 * K, 0, 1);
        v8f acc0 = {}, acc1 = {};
        for (int kk = 0; kk < K; kk += 32) {
            AFrag af0, af1;
            af0.q[0] = *(const uint4*)(Arow0 + kk + koffA);
            af0.q[1] = *(const uint4*)(Arow0 + kk + 8 + koffA);
            af1.q[0] = *(const uint4*)(Arow1 + kk + koffA);
            af1.q[1] = *(const uint4*)(Arow1 + kk + 8 + koffA);
            v16bf b = *(const v16bf*)(Bn + kk + koffB);
            acc0 = WMMA_BF16(af0.v, b, acc0);
            acc1 = WMMA_BF16(af1.v, b, acc1);
        }
        float bv = bias[n];
        #pragma unroll
        for (int r = 0; r < 8; ++r) {
            float v0 = acc0[r] + bv;
            float v1 = acc1[r] + bv;
            if (ACT == 1) { v0 = lrelu(v0); v1 = lrelu(v1); }
            C[(size_t)(rowb + mbase + r) * H_DIM + n]      = f2bf(v0);
            C[(size_t)(rowb + 16 + mbase + r) * H_DIM + n] = f2bf(v1);
        }
    }
}

// ---------------------------------------------------------------------------
// Final: per-variant head reduction, conf scaling, tanh clamp
__global__ void final_kernel(const unsigned short* __restrict__ O,  // 3 x V x 256
                             const float* __restrict__ W2,          // 3 x 256
                             const float* __restrict__ b2,          // 3
                             const int*   __restrict__ vtypes,      // V
                             const float* __restrict__ alt_cnt,     // V
                             const float* __restrict__ conf,        // 11
                             const float* __restrict__ max_logit_p, // 1
                             float* __restrict__ out)               // V
{
    int v = blockIdx.x * blockDim.x + threadIdx.x;
    if (v >= V_SEG) return;
    int t = vtypes[v];
    const unsigned short* o = O + ((size_t)t * V_SEG + v) * H_DIM;
    const float* w2 = W2 + t * H_DIM;
    float s = b2[t];
    for (int e = 0; e < H_DIM; ++e) s += bf2f(o[e]) * w2[e];
    int c = (int)alt_cnt[v]; if (c > MAX_ALT) c = MAX_ALT;
    float logit = s * conf[c];
    float ml = max_logit_p[0];
    out[v] = ml * tanhf(logit / ml);
}

// ---------------------------------------------------------------------------
extern "C" void kernel_launch(void* const* d_in, const int* in_sizes, int n_in,
                              void* d_out, int out_size, void* d_ws, size_t ws_size,
                              hipStream_t stream) {
    const float* reads    = (const float*)d_in[0];
    const float* info     = (const float*)d_in[1];
    const int*   ref_seg  = (const int*)  d_in[2];
    const int*   alt_seg  = (const int*)  d_in[3];
    const int*   vtypes   = (const int*)  d_in[4];
    const float* phi_W0   = (const float*)d_in[5];
    const float* phi_b0   = (const float*)d_in[6];
    const float* phi_W1   = (const float*)d_in[7];
    const float* phi_b1   = (const float*)d_in[8];
    const float* om_W0    = (const float*)d_in[9];
    const float* om_b0    = (const float*)d_in[10];
    const float* rho_W0   = (const float*)d_in[11];
    const float* rho_b0   = (const float*)d_in[12];
    const float* rho_W1   = (const float*)d_in[13];
    const float* rho_b1   = (const float*)d_in[14];
    const float* out_W1   = (const float*)d_in[15];
    const float* out_b1   = (const float*)d_in[16];
    const float* out_W2   = (const float*)d_in[17];
    const float* out_b2   = (const float*)d_in[18];
    const float* conf     = (const float*)d_in[19];
    const float* maxlog   = (const float*)d_in[20];
    float* out = (float*)d_out;

    // ---- workspace layout (bytes) ----
    char* ws = (char*)d_ws;
    unsigned short* Wt_phi1 = (unsigned short*)(ws + 0);          // 256x256
    unsigned short* Wt_rho0 = (unsigned short*)(ws + 131072);     // 256x768
    unsigned short* Wt_rho1 = (unsigned short*)(ws + 524288);     // 256x256
    unsigned short* Wt_out1 = (unsigned short*)(ws + 655360);     // 3 x 256x256
    float* ref_sums = (float*)(ws + 1048576);                     // V x 256
    float* alt_sums = (float*)(ws + 9437184);                     // V x 256
    float* ref_cnt  = (float*)(ws + 17825792);                    // V
    float* alt_cnt  = (float*)(ws + 17858560);                    // V
    unsigned short* X   = (unsigned short*)(ws + 17891328);       // V x 768
    unsigned short* Y1  = (unsigned short*)(ws + 30474240);       // V x 256
    unsigned short* AGG = (unsigned short*)(ws + 34668544);       // V x 256
    unsigned short* O   = (unsigned short*)(ws + 38862848);       // 3 x V x 256

    // 1. zero accumulators (sums + counts are contiguous)
    int nzero = V_SEG * H_DIM * 2 + V_SEG * 2;
    zero_kernel<<<1024, 256, 0, stream>>>(ref_sums, nzero);

    // 2. weight transpose/convert to bf16 N-major
    tw_kernel<<<(H_DIM * H_DIM + 255) / 256, 256, 0, stream>>>(phi_W1, Wt_phi1, H_DIM, H_DIM);
    tw_kernel<<<(3 * H_DIM * H_DIM + 255) / 256, 256, 0, stream>>>(rho_W0, Wt_rho0, 3 * H_DIM, H_DIM);
    tw_kernel<<<(H_DIM * H_DIM + 255) / 256, 256, 0, stream>>>(rho_W1, Wt_rho1, H_DIM, H_DIM);
    for (int t = 0; t < 3; ++t)
        tw_kernel<<<(H_DIM * H_DIM + 255) / 256, 256, 0, stream>>>(
            out_W1 + t * H_DIM * H_DIM, Wt_out1 + t * H_DIM * H_DIM, H_DIM, H_DIM);

    // 3. segment counts
    count_kernel<<<(NREF + 255) / 256, 256, 0, stream>>>(ref_seg, ref_cnt, NREF);
    count_kernel<<<(NALT + 255) / 256, 256, 0, stream>>>(alt_seg, alt_cnt, NALT);

    // 4. fused phi MLP + segmented sum (the 51.5 GFLOP kernel)
    size_t phi_smem = (size_t)PHI_ROWS * H_DIM * 2        // hbuf
                    + (size_t)PHI_ROWS * F_READ * 4       // rbuf
                    + (size_t)PHI_ROWS * 4 * 2;           // segbuf + refbuf
    phi_kernel<<<NTOT / PHI_ROWS, 128, phi_smem, stream>>>(
        reads, ref_seg, alt_seg, phi_W0, phi_b0, Wt_phi1, phi_b1, ref_sums, alt_sums);

    // 5. concat input
    buildx_kernel<<<(V_SEG * H_DIM + 255) / 256, 256, 0, stream>>>(
        ref_sums, alt_sums, ref_cnt, alt_cnt, info, om_W0, om_b0, X);

    // 6. rho MLP
    gemm_kernel<1><<<V_SEG / 256, 256, 0, stream>>>(X, Wt_rho0, rho_b0, Y1, 3 * H_DIM);
    gemm_kernel<0><<<V_SEG / 256, 256, 0, stream>>>(Y1, Wt_rho1, rho_b1, AGG, H_DIM);

    // 7. output heads
    for (int t = 0; t < 3; ++t)
        gemm_kernel<1><<<V_SEG / 256, 256, 0, stream>>>(
            AGG, Wt_out1 + t * H_DIM * H_DIM, out_b1 + t * H_DIM,
            O + (size_t)t * V_SEG * H_DIM, H_DIM);

    // 8. final reduction + select + conf + tanh clamp
    final_kernel<<<(V_SEG + 255) / 256, 256, 0, stream>>>(
        O, out_W2, out_b2, vtypes, alt_cnt, conf, maxlog, out);
}
